// GCN_width_69277822484763
// MI455X (gfx1250) — matrix operand order repaired and checked by reading, not compiled, not run on hardware
//
#include <hip/hip_runtime.h>
#include <hip/hip_bf16.h>

typedef __attribute__((ext_vector_type(16))) __bf16 v16bf;
typedef __attribute__((ext_vector_type(8)))  __bf16 v8bf;
typedef __attribute__((ext_vector_type(8)))  float  v8f;
typedef __attribute__((ext_vector_type(4)))  float  v4f;
typedef __attribute__((ext_vector_type(4)))  unsigned int u32x4;
typedef __attribute__((ext_vector_type(4)))  int i32x4;
typedef __attribute__((ext_vector_type(8)))  int i32x8;

#define HAVE_TDM __has_builtin(__builtin_amdgcn_tensor_load_to_lds)

static inline unsigned div_up_u(long long a, long long b) { return (unsigned)((a + b - 1) / b); }

constexpr int ilog2c(int v) { int r = 0; while (v > 1) { v >>= 1; ++r; } return r; }

// ---------------------------------------------------------------------------
// Split fp32 W[K x NCOL] into bf16 hi/lo, transposed to [NCOL x K] so that the
// per-lane WMMA B-fragment (16 consecutive K for one column) is a contiguous
// 32-byte run.
// ---------------------------------------------------------------------------
__global__ void prep_w_kernel(const float* __restrict__ W, __bf16* __restrict__ Whi,
                              __bf16* __restrict__ Wlo, int K, int NCOL) {
  int idx = blockIdx.x * blockDim.x + threadIdx.x;
  if (idx >= K * NCOL) return;
  int k = idx / NCOL, c = idx % NCOL;
  float w  = W[idx];
  __bf16 hi = (__bf16)w;
  float  r  = w - (float)hi;
  __bf16 lo = (__bf16)r;
  Whi[(size_t)c * K + k] = hi;
  Wlo[(size_t)c * K + k] = lo;
}

// ---------------------------------------------------------------------------
// Degree / symmetric norm (self loops included: deg starts at 1)
// ---------------------------------------------------------------------------
__global__ void deg_init_kernel(float* __restrict__ deg, long long N) {
  long long i = (long long)blockIdx.x * blockDim.x + threadIdx.x;
  if (i < N) deg[i] = 1.0f;
}
__global__ void deg_edge_kernel(const long long* __restrict__ dst, float* __restrict__ deg,
                                long long E) {
  long long e = (long long)blockIdx.x * blockDim.x + threadIdx.x;
  if (e < E) atomicAdd(&deg[dst[e]], 1.0f);
}
__global__ void deg_rsqrt_kernel(float* __restrict__ deg, long long N) {
  long long i = (long long)blockIdx.x * blockDim.x + threadIdx.x;
  if (i < N) {
    float d = deg[i];
    deg[i] = d > 0.0f ? rsqrtf(d) : 0.0f;   // deg -> dinv in place
  }
}

#if HAVE_TDM
// ---------------------------------------------------------------------------
// TDM: DMA a [nrows x row_elems] bf16 tensor (contiguous rows) into LDS,
// inserting pad_amount DWORDs of LDS padding after each row (bank skew).
// D# bit layout per CDNA5 ISA 8.3/8.4. 6-arg builtin form (clang-23 style):
// (g0:u32x4, g1:i32x8, g2:i32x4, g3:i32x4, extra:i32x8, cpol:i32).
// ---------------------------------------------------------------------------
__device__ __forceinline__ void tdm_load_2d(const void* gsrc, unsigned lds_byte_addr,
                                            int row_elems, int nrows,
                                            int pad_interval, int pad_amount) {
  unsigned long long ga = (unsigned long long)(size_t)gsrc;
  u32x4 g0;
  g0[0] = 1u;                                                   // count=1 (valid, user mode)
  g0[1] = lds_byte_addr;                                        // lds_addr (bytes)
  g0[2] = (unsigned)ga;                                         // global_addr[31:0]
  g0[3] = (unsigned)((ga >> 32) & 0x01ffffffu) | (2u << 30);    // global_addr[56:32] | type=2
  unsigned d0 = (1u << 16)                                      // data_size = 1 -> 2 bytes
              | (1u << 20)                                      // pad_enable
              | ((unsigned)pad_interval << 22)                  // pad_interval code
              | ((unsigned)pad_amount  << 25);                  // pad_amount code
  i32x8 g1;
  g1[0] = (int)d0;                                              // workgroup_mask=0
  g1[1] = (int)(((unsigned)row_elems & 0xffffu) << 16);         // tensor_dim0[15:0]
  g1[2] = (int)((((unsigned)row_elems >> 16) & 0xffffu) |
                (((unsigned)nrows & 0xffffu) << 16));           // dim0[31:16] | dim1[15:0]
  g1[3] = (int)((((unsigned)nrows >> 16) & 0xffffu) |
                (((unsigned)row_elems & 0xffffu) << 16));       // dim1[31:16] | tile_dim0
  g1[4] = (int)((unsigned)nrows & 0xffffu);                     // tile_dim1 (tile_dim2 = 0)
  g1[5] = row_elems;                                            // tensor_dim0_stride[31:0]
  g1[6] = 0;
  g1[7] = 0;
  i32x4 z4 = {0, 0, 0, 0};
  i32x8 z8 = {0, 0, 0, 0, 0, 0, 0, 0};
  __builtin_amdgcn_tensor_load_to_lds(g0, g1, z4, z4, z8, 0);
}
#endif

// ---------------------------------------------------------------------------
// bf16x3 split-precision GEMM: Hout[M x NCOL] = act(A[M x K]) @ W[K x NCOL]
// One wave per 16-row tile, covering all NCOL columns (A read once from HBM,
// non-temporal). Split weights are staged in LDS via the Tensor Data Mover
// with a 16B/row skew, then consumed as conflict-free ds_load_b128 pairs.
// ---------------------------------------------------------------------------
template<int K, int NCOL, bool RELU>
__global__ void __launch_bounds__(256)
gemm_bf16x3_kernel(const float* __restrict__ A, const __bf16* __restrict__ Whi,
                   const __bf16* __restrict__ Wlo, float* __restrict__ Hout, int M) {
  constexpr int NT    = NCOL / 16;
  constexpr int PITCH = K * 2 + 16;          // LDS bytes per weight row (+16B skew)
  constexpr int LDSB  = NCOL * PITCH;        // bytes per (hi|lo) plane
  __shared__ unsigned char smem[2 * LDSB];

  const int lane = threadIdx.x & 31;
  const int wv   = threadIdx.x >> 5;
  const int half = lane >> 4;
  const int l16  = lane & 15;

  // ---- stage split weights into LDS ----
#if HAVE_TDM
  if (wv == 0) {                             // wave-uniform; TDM ignores EXEC
    unsigned lds0 = (unsigned)(size_t)(void*)&smem[0];
    constexpr int ROW_DW = (K * 2) / 4;      // dwords per weight row
    constexpr int PAD_IV = ilog2c(ROW_DW) - 1;  // 2^(code+1) dwords between pads
    constexpr int PAD_AM = 3;                // code 3 -> 4 dwords = 16 bytes
    tdm_load_2d(Whi, lds0,        K, NCOL, PAD_IV, PAD_AM);
    tdm_load_2d(Wlo, lds0 + LDSB, K, NCOL, PAD_IV, PAD_AM);
    __builtin_amdgcn_s_wait_tensorcnt(0);
  }
#else
  for (int i = threadIdx.x; i < K * NCOL; i += (int)blockDim.x) {
    int c = i / K, k = i - c * K;
    *(__bf16*)(&smem[(size_t)c * PITCH + k * 2])        = Whi[(size_t)c * K + k];
    *(__bf16*)(&smem[LDSB + (size_t)c * PITCH + k * 2]) = Wlo[(size_t)c * K + k];
  }
#endif
  __syncthreads();

  int wave = blockIdx.x * (blockDim.x >> 5) + wv;
  if (wave * 16 >= M) return;                // wave-uniform: EXEC stays all-ones

  const float* arow = A + (size_t)(wave * 16 + l16) * K;

  v8f acc[NT] = {};

  for (int k0 = 0; k0 < K; k0 += 32) {
    // ---- load + split the A fragment (single-use data: non-temporal) ----
    // A k order used here (contiguous, not the ISA table order): slot j = K=k0+half*16+j (matches B).
    v16bf a_hi, a_lo;
#pragma unroll
    for (int g = 0; g < 2; ++g) {
      const float* p = arow + k0 + half * 16 + g * 8;
      v4f f0 = __builtin_nontemporal_load((const v4f*)p);
      v4f f1 = __builtin_nontemporal_load((const v4f*)(p + 4));
#pragma unroll
      for (int j = 0; j < 8; ++j) {
        float v = (j < 4) ? f0[j] : f1[j - 4];
        if (RELU) v = fmaxf(v, 0.0f);
        __bf16 hi = (__bf16)v;
        float  r  = v - (float)hi;
        a_hi[g * 8 + j] = hi;
        a_lo[g * 8 + j] = (__bf16)r;
      }
    }
    // ---- per N-tile: B fragments from LDS (skewed pitch) + 3 WMMAs ----
    // B k order used here (contiguous, same as A): per lane col=t*16+l16, slot j = K=k0+half*16+j.
#pragma unroll
    for (int t = 0; t < NT; ++t) {
      unsigned boff = (unsigned)(t * 16 + l16) * PITCH + (unsigned)(k0 + half * 16) * 2;
      v8bf h0 = *(const v8bf*)(&smem[boff]);
      v8bf h1 = *(const v8bf*)(&smem[boff + 16]);
      v8bf l0 = *(const v8bf*)(&smem[LDSB + boff]);
      v8bf l1 = *(const v8bf*)(&smem[LDSB + boff + 16]);
      v16bf b_hi = __builtin_shufflevector(h0, h1, 0,1,2,3,4,5,6,7,8,9,10,11,12,13,14,15);
      v16bf b_lo = __builtin_shufflevector(l0, l1, 0,1,2,3,4,5,6,7,8,9,10,11,12,13,14,15);
      v8f c = acc[t];
      c = __builtin_amdgcn_wmma_f32_16x16x32_bf16(false, a_hi, false, b_hi, (short)0, c, false, false);
      c = __builtin_amdgcn_wmma_f32_16x16x32_bf16(false, a_hi, false, b_lo, (short)0, c, false, false);
      c = __builtin_amdgcn_wmma_f32_16x16x32_bf16(false, a_lo, false, b_hi, (short)0, c, false, false);
      acc[t] = c;
    }
  }

  // ---- store C: VGPR r, lanes 0-15 -> (M=r, N=l16); lanes 16-31 -> (M=r+8) ----
#pragma unroll
  for (int t = 0; t < NT; ++t) {
    int col = t * 16 + l16;
#pragma unroll
    for (int r = 0; r < 8; ++r) {
      int row = wave * 16 + half * 8 + r;
      Hout[(size_t)row * NCOL + col] = acc[t][r];
    }
  }
}

// ---------------------------------------------------------------------------
// out[i,f] = bias[f] + h[i,f] * dinv[i]^2   (self-loop term + bias)
// ---------------------------------------------------------------------------
template<int LOGF>
__global__ void self_init_kernel(const float* __restrict__ h, const float* __restrict__ dinv,
                                 const float* __restrict__ bias, float* __restrict__ out,
                                 long long N) {
  long long idx = (long long)blockIdx.x * blockDim.x + threadIdx.x;
  if (idx >= (N << LOGF)) return;
  long long i = idx >> LOGF;
  int f = (int)(idx & ((1 << LOGF) - 1));
  float di = dinv[i];
  out[idx] = bias[f] + h[idx] * di * di;
}

// ---------------------------------------------------------------------------
// out[dst,f] += h[src,f] * dinv[src] * dinv[dst]   (L2-resident f32 atomics)
// ---------------------------------------------------------------------------
template<int LOGF>
__global__ void scatter_kernel(const long long* __restrict__ src, const long long* __restrict__ dst,
                               const float* __restrict__ dinv, const float* __restrict__ h,
                               float* __restrict__ out, long long E) {
  long long idx = (long long)blockIdx.x * blockDim.x + threadIdx.x;
  if (idx >= (E << LOGF)) return;
  long long e = idx >> LOGF;
  int f = (int)(idx & ((1 << LOGF) - 1));
  long long s = src[e], d = dst[e];
  float nrm = dinv[s] * dinv[d];
  atomicAdd(out + ((d << LOGF) + f), h[(s << LOGF) + f] * nrm);
}

// ---------------------------------------------------------------------------
// Row-wise log_softmax over 64 classes: one wave per row, 2 elems/lane.
// ---------------------------------------------------------------------------
__global__ void log_softmax64_kernel(float* __restrict__ out, long long N) {
  long long row = (long long)blockIdx.x * (blockDim.x >> 5) + (threadIdx.x >> 5);
  if (row >= N) return;
  int lane = threadIdx.x & 31;
  float* p = out + (row << 6);
  float a = p[lane];
  float b = p[lane + 32];
  float m = fmaxf(a, b);
#pragma unroll
  for (int off = 16; off; off >>= 1) m = fmaxf(m, __shfl_xor(m, off, 32));
  float s = expf(a - m) + expf(b - m);
#pragma unroll
  for (int off = 16; off; off >>= 1) s += __shfl_xor(s, off, 32);
  float lse = m + logf(s);
  p[lane]      = a - lse;
  p[lane + 32] = b - lse;
}

// ---------------------------------------------------------------------------
extern "C" void kernel_launch(void* const* d_in, const int* in_sizes, int n_in,
                              void* d_out, int out_size, void* d_ws, size_t ws_size,
                              hipStream_t stream) {
  const int FIN = 512, H = 128, C = 64;

  const float*     x   = (const float*)d_in[0];
  const long long* ei  = (const long long*)d_in[1];
  const float*     W1  = (const float*)d_in[2];
  const float*     b1  = (const float*)d_in[3];
  const float*     W2  = (const float*)d_in[4];
  const float*     b2  = (const float*)d_in[5];
  float* out = (float*)d_out;

  long long N = (long long)in_sizes[0] / FIN;
  long long E = (long long)in_sizes[1] / 2;
  const long long* src = ei;
  const long long* dst = ei + E;

  // workspace carve-out (256B aligned)
  char* ws = (char*)d_ws;
  size_t off = 0;
  auto alloc = [&](size_t bytes) {
    char* p = ws + off;
    off += (bytes + 255) & ~(size_t)255;
    return p;
  };
  float*  dinv = (float*)alloc((size_t)N * 4);
  float*  h    = (float*)alloc((size_t)N * H * 4);   // layer-1 transform; reused as h2
  float*  agg1 = (float*)alloc((size_t)N * H * 4);   // layer-1 aggregated output
  __bf16* w1hi = (__bf16*)alloc((size_t)FIN * H * 2);
  __bf16* w1lo = (__bf16*)alloc((size_t)FIN * H * 2);
  __bf16* w2hi = (__bf16*)alloc((size_t)H * C * 2);
  __bf16* w2lo = (__bf16*)alloc((size_t)H * C * 2);
  (void)ws_size; (void)n_in; (void)out_size;

  const int TPB = 256;

  // weight split/transpose
  prep_w_kernel<<<div_up_u((long long)FIN * H, TPB), TPB, 0, stream>>>(W1, w1hi, w1lo, FIN, H);
  prep_w_kernel<<<div_up_u((long long)H * C,  TPB), TPB, 0, stream>>>(W2, w2hi, w2lo, H, C);

  // degrees -> dinv
  deg_init_kernel<<<div_up_u(N, TPB), TPB, 0, stream>>>(dinv, N);
  deg_edge_kernel<<<div_up_u(E, TPB), TPB, 0, stream>>>(dst, dinv, E);
  deg_rsqrt_kernel<<<div_up_u(N, TPB), TPB, 0, stream>>>(dinv, N);

  unsigned gemm_blocks = div_up_u((N + 15) / 16, TPB / 32);

  // ---- layer 1 ----
  gemm_bf16x3_kernel<512, 128, false><<<gemm_blocks, TPB, 0, stream>>>(x, w1hi, w1lo, h, (int)N);
  self_init_kernel<7><<<div_up_u(N << 7, TPB), TPB, 0, stream>>>(h, dinv, b1, agg1, N);
  scatter_kernel<7><<<div_up_u(E << 7, TPB), TPB, 0, stream>>>(src, dst, dinv, h, agg1, E);

  // ---- layer 2 (ReLU folded into A-load; h reused as h2) ----
  gemm_bf16x3_kernel<128, 64, true><<<gemm_blocks, TPB, 0, stream>>>(agg1, w2hi, w2lo, h, (int)N);
  self_init_kernel<6><<<div_up_u(N << 6, TPB), TPB, 0, stream>>>(h, dinv, b2, out, N);
  scatter_kernel<6><<<div_up_u(E << 6, TPB), TPB, 0, stream>>>(src, dst, dinv, h, out, E);

  // ---- log_softmax ----
  log_softmax64_kernel<<<div_up_u(N, TPB / 32), TPB, 0, stream>>>(out, N);
}
